// MultiheadAttention_62878321214047
// MI455X (gfx1250) — compile-verified, matrix operand order checked by reading, not run on hardware
//
#include <hip/hip_runtime.h>
#include <hip/hip_bf16.h>

typedef __attribute__((ext_vector_type(16))) _Float16 v16h;
typedef __attribute__((ext_vector_type(8)))  float    v8f;

#define B_    4
#define N1_   2048
#define N2_   1024
#define IND_  768
#define OUTD_ 1024
#define H_    16
#define DH_   64

#define BM 64
#define BN 64
#define KT 32
#define LPAD 40   // halves per LDS row: 80B stride -> conflict-free across 16 rows

// ---- helpers -------------------------------------------------------------

__device__ __forceinline__ v16h ld16h(const _Float16* p) {
  union { uint4 u[2]; v16h h; } r;
  r.u[0] = *(const uint4*)(p);
  r.u[1] = *(const uint4*)(p + 8);
  return r.h;
}

__device__ __forceinline__ v8f wmma_f16(v16h a, v16h b, v8f c) {
  // D = A(16x32 f16) * B(32x16 f16) + C(16x16 f32)
  return __builtin_amdgcn_wmma_f32_16x16x32_f16(
      /*neg_a=*/false, a, /*neg_b=*/false, b,
      /*c_mod=*/(short)0, c, /*reuse_a=*/false, /*reuse_b=*/false);
}

// Async copy 16B memory -> LDS (GLOBAL_LOAD_ASYNC_TO_LDS_B128, ASYNCcnt).
// LDS byte address = low 32 bits of the generic pointer (ISA 10.2: generic
// pointers to LDS carry the LDS offset in addr[31:0]).
__device__ __forceinline__ void async_copy_b128(const _Float16* g, _Float16* l) {
  unsigned int lds = (unsigned int)(unsigned long long)l;
  asm volatile("global_load_async_to_lds_b128 %0, %1, off"
               :: "v"(lds), "v"(g)
               : "memory");
}

__device__ __forceinline__ void wait_async0() {
  asm volatile("s_wait_asynccnt 0x0" ::: "memory");
}

// ---- f32 -> f16 weight conversion ---------------------------------------

__global__ __launch_bounds__(256)
void f32_to_f16_kernel(const float* __restrict__ x, _Float16* __restrict__ y, int n) {
  int i = blockIdx.x * 256 + threadIdx.x;
  if (i < n) y[i] = (_Float16)x[i];
}

// ---- LayerNorm (f32 in, f16 out) ----------------------------------------

__global__ __launch_bounds__(256)
void layernorm_kernel(const float* __restrict__ X, const float* __restrict__ g,
                      const float* __restrict__ bt, _Float16* __restrict__ Y, int dim) {
  __shared__ float red[256];
  const int tid = threadIdx.x;
  const float* row = X + (size_t)blockIdx.x * dim;
  _Float16* yrow = Y + (size_t)blockIdx.x * dim;

  float s = 0.f, ss = 0.f;
  for (int i = tid; i < dim; i += 256) { float x = row[i]; s += x; ss += x * x; }
  red[tid] = s; __syncthreads();
  for (int k = 128; k > 0; k >>= 1) { if (tid < k) red[tid] += red[tid + k]; __syncthreads(); }
  s = red[0]; __syncthreads();
  red[tid] = ss; __syncthreads();
  for (int k = 128; k > 0; k >>= 1) { if (tid < k) red[tid] += red[tid + k]; __syncthreads(); }
  ss = red[0];

  const float mu  = s / (float)dim;
  const float var = ss / (float)dim - mu * mu;
  const float inv = rsqrtf(var + 1e-5f);
  for (int i = tid; i < dim; i += 256)
    yrow[i] = (_Float16)((row[i] - mu) * inv * g[i] + bt[i]);
}

// ---- Tiled GEMM  C[M,Nn] = A[M,Kd] * W[Nn,Kd]^T + bias --------------------
// Double-buffered LDS staged with async global->LDS copies (ASYNCcnt).
// MODE 0: store f16 row-major [M][Nn]           (Q, K projections)
// MODE 1: store f16 transposed [B][H][DH][ntok] (V projection)
// MODE 2: store f32 row-major + residual        (output projection)

template<int MODE>
__global__ __launch_bounds__(256)
void gemm_bias_kernel(const _Float16* __restrict__ A, const _Float16* __restrict__ W,
                      const float* __restrict__ bias, const float* __restrict__ resid,
                      _Float16* __restrict__ outh, float* __restrict__ outf,
                      int M, int Kd, int Nn, int ntok) {
  __shared__ _Float16 lA[2][BM][LPAD];
  __shared__ _Float16 lB[2][BN][LPAD];

  const int tid  = threadIdx.x;
  const int lane = tid & 31;
  const int wave = tid >> 5;
  const int m0 = blockIdx.x * BM;
  const int n0 = blockIdx.y * BN;

  const int rowL = tid >> 2;   // 0..63
  const int segL = tid & 3;    // 16B chunk within 64B row

  const int wm = (wave & 3) * 16;  // wave M sub-tile
  const int wn = (wave >> 2) * 32; // wave N sub-tile (two 16-wide WMMA tiles)

  const int nk = Kd / KT;

  const _Float16* gA = A + (size_t)(m0 + rowL) * Kd + segL * 8;
  const _Float16* gB = W + (size_t)(n0 + rowL) * Kd + segL * 8;

  // stage tile 0 (async DMA to LDS)
  async_copy_b128(gA, &lA[0][rowL][segL * 8]);
  async_copy_b128(gB, &lB[0][rowL][segL * 8]);
  wait_async0();
  __syncthreads();

  v8f c0 = {};
  v8f c1 = {};
  const int fr = lane & 15;          // row (A) / col (B) within tile
  const int fg = (lane >> 4) * 16;   // K-half group

  for (int kt = 0; kt < nk; ++kt) {
    const int cur = kt & 1;
    if (kt + 2 < nk) {  // prefetch two tiles ahead (global_prefetch)
      __builtin_prefetch(gA + (kt + 2) * KT, 0, 0);
      __builtin_prefetch(gB + (kt + 2) * KT, 0, 0);
    }
    if (kt + 1 < nk) {  // async-stage next K tile into the other buffer
      const int nxt = cur ^ 1;
      const int kb  = (kt + 1) * KT;
      async_copy_b128(gA + kb, &lA[nxt][rowL][segL * 8]);
      async_copy_b128(gB + kb, &lB[nxt][rowL][segL * 8]);
    }
    v16h a  = ld16h(&lA[cur][wm + fr][fg]);
    v16h b0 = ld16h(&lB[cur][wn + fr][fg]);
    v16h b1 = ld16h(&lB[cur][wn + 16 + fr][fg]);
    c0 = wmma_f16(a, b0, c0);
    c1 = wmma_f16(a, b1, c1);
    wait_async0();     // next buffer fully landed in LDS
    __syncthreads();
  }

  // epilogue: accumulator VGPR r holds M = r (lanes 0-15) / 8+r (lanes 16-31), N = lane%16
  const int hiM = (lane >> 4) * 8;
  const float bv0 = bias[n0 + wn + fr];
  const float bv1 = bias[n0 + wn + 16 + fr];
#pragma unroll
  for (int r = 0; r < 8; ++r) {
    const int mg  = m0 + wm + hiM + r;
    const int ng0 = n0 + wn + fr;
    const int ng1 = ng0 + 16;
    const float x0 = c0[r] + bv0;
    const float x1 = c1[r] + bv1;
    if (MODE == 0) {
      outh[(size_t)mg * Nn + ng0] = (_Float16)x0;
      outh[(size_t)mg * Nn + ng1] = (_Float16)x1;
    } else if (MODE == 1) {
      const int b = mg / ntok, tok = mg % ntok;
      const int h0 = ng0 >> 6, d0 = ng0 & 63;
      const int h1 = ng1 >> 6, d1 = ng1 & 63;
      outh[(((size_t)b * H_ + h0) * DH_ + d0) * ntok + tok] = (_Float16)x0;
      outh[(((size_t)b * H_ + h1) * DH_ + d1) * ntok + tok] = (_Float16)x1;
    } else {
      outf[(size_t)mg * Nn + ng0] = resid[(size_t)mg * Nn + ng0] + x0;
      outf[(size_t)mg * Nn + ng1] = resid[(size_t)mg * Nn + ng1] + x1;
    }
  }
}

// ---- scores = Q K^T / 8 : direct-from-global WMMA ------------------------
// Q: [B][N2][H][DH] f16, K: [B][N1][H][DH] f16, S: [B*H][N2][N1] f32

__global__ __launch_bounds__(256)
void attn_scores_kernel(const _Float16* __restrict__ Q, const _Float16* __restrict__ K,
                        float* __restrict__ S) {
  const int tid = threadIdx.x, lane = tid & 31, wave = tid >> 5;
  const int q0 = blockIdx.x * 16;
  const int k0 = blockIdx.y * 512 + wave * 64;
  const int bh = blockIdx.z;
  const int b = bh >> 4, h = bh & 15;
  const int fr = lane & 15, fg = (lane >> 4) * 16;

  const _Float16* qp = Q + (((size_t)(b * N2_ + q0 + fr)) * H_ + h) * DH_ + fg;
  const v16h a0 = ld16h(qp);        // d = 0..31
  const v16h a1 = ld16h(qp + 32);   // d = 32..63

  float* srow = S + ((size_t)bh * N2_ + q0) * N1_;
  const int hiM = (lane >> 4) * 8;

#pragma unroll
  for (int t = 0; t < 4; ++t) {
    const int kt0 = k0 + t * 16;
    const _Float16* kp = K + (((size_t)(b * N1_ + kt0 + fr)) * H_ + h) * DH_ + fg;
    const v16h b0 = ld16h(kp);
    const v16h b1 = ld16h(kp + 32);
    v8f c = {};
    c = wmma_f16(a0, b0, c);
    c = wmma_f16(a1, b1, c);
#pragma unroll
    for (int r = 0; r < 8; ++r)
      srow[(size_t)(hiM + r) * N1_ + kt0 + fr] = c[r] * 0.125f;
  }
}

// ---- row softmax over N1, in place in d_out ------------------------------

__global__ __launch_bounds__(256)
void softmax_kernel(float* __restrict__ S) {
  __shared__ float red[256];
  const int tid = threadIdx.x;
  float* row = S + (size_t)blockIdx.x * N1_;
  float v[8];
  float m = -3.402823466e38f;
#pragma unroll
  for (int i = 0; i < 8; ++i) { v[i] = row[tid + i * 256]; m = fmaxf(m, v[i]); }
  red[tid] = m; __syncthreads();
  for (int s = 128; s > 0; s >>= 1) { if (tid < s) red[tid] = fmaxf(red[tid], red[tid + s]); __syncthreads(); }
  m = red[0]; __syncthreads();
  float sum = 0.f;
#pragma unroll
  for (int i = 0; i < 8; ++i) { v[i] = __expf(v[i] - m); sum += v[i]; }
  red[tid] = sum; __syncthreads();
  for (int s = 128; s > 0; s >>= 1) { if (tid < s) red[tid] += red[tid + s]; __syncthreads(); }
  const float inv = 1.0f / red[0];
#pragma unroll
  for (int i = 0; i < 8; ++i) row[tid + i * 256] = v[i] * inv;
}

// ---- ctrl = P V : P f32 from d_out (convert in-register), Vt [B][H][DH][N1]

__global__ __launch_bounds__(256)
void attn_av_kernel(const float* __restrict__ P, const _Float16* __restrict__ Vt,
                    _Float16* __restrict__ ctrl) {
  const int tid = threadIdx.x, lane = tid & 31, wave = tid >> 5;
  const int bh = blockIdx.z, b = bh >> 4, h = bh & 15;
  const int q0 = blockIdx.x * 128 + wave * 16;
  const int fr = lane & 15, fg = (lane >> 4) * 16;

  v8f c[4] = {{}, {}, {}, {}};
  const float* prow = P + ((size_t)bh * N2_ + q0 + fr) * N1_;
  const _Float16* vbase = Vt + (size_t)bh * DH_ * N1_;

  for (int kk = 0; kk < N1_; kk += 32) {
    const float* pp = prow + kk + fg;  // 16 contiguous f32 probs
    v16h a;
#pragma unroll
    for (int i = 0; i < 16; ++i) a[i] = (_Float16)pp[i];
#pragma unroll
    for (int nt = 0; nt < 4; ++nt) {
      const _Float16* vp = vbase + (size_t)(nt * 16 + fr) * N1_ + kk + fg;
      c[nt] = wmma_f16(a, ld16h(vp), c[nt]);
    }
  }

  const int hiM = (lane >> 4) * 8;
#pragma unroll
  for (int nt = 0; nt < 4; ++nt)
#pragma unroll
    for (int r = 0; r < 8; ++r) {
      const int q = q0 + hiM + r;
      const int d = nt * 16 + fr;
      ctrl[(size_t)(b * N2_ + q) * OUTD_ + h * DH_ + d] = (_Float16)c[nt][r];
    }
}

// ---- host launch ---------------------------------------------------------

extern "C" void kernel_launch(void* const* d_in, const int* in_sizes, int n_in,
                              void* d_out, int out_size, void* d_ws, size_t ws_size,
                              hipStream_t stream) {
  const float* in_x     = (const float*)d_in[0];   // [B,N1,IN]
  const float* out_x    = (const float*)d_in[1];   // [B,N2,OUT]
  const float* q_w      = (const float*)d_in[2];
  const float* q_b      = (const float*)d_in[3];
  const float* k_w      = (const float*)d_in[4];
  const float* k_b      = (const float*)d_in[5];
  const float* v_w      = (const float*)d_in[6];
  const float* v_b      = (const float*)d_in[7];
  const float* proj_w   = (const float*)d_in[8];
  const float* proj_b   = (const float*)d_in[9];
  const float* ln_in_g  = (const float*)d_in[10];
  const float* ln_in_b  = (const float*)d_in[11];
  const float* ln_out_g = (const float*)d_in[12];
  const float* ln_out_b = (const float*)d_in[13];

  char* ws = (char*)d_ws;
  size_t off = 0;
  auto carve = [&](size_t elems) -> _Float16* {
    _Float16* p = (_Float16*)(ws + off);
    off += ((elems * sizeof(_Float16)) + 255) & ~(size_t)255;
    return p;
  };
  _Float16* ln_in_h  = carve((size_t)B_ * N1_ * IND_);    // 12.6 MB
  _Float16* ln_out_h = carve((size_t)B_ * N2_ * OUTD_);   //  8.4 MB
  _Float16* q_h      = carve((size_t)B_ * N2_ * OUTD_);   //  8.4 MB
  _Float16* k_h      = carve((size_t)B_ * N1_ * OUTD_);   // 16.8 MB
  _Float16* vt_h     = carve((size_t)B_ * N1_ * OUTD_);   // 16.8 MB (transposed [B][H][DH][N1])
  _Float16* ctrl_h   = carve((size_t)B_ * N2_ * OUTD_);   //  8.4 MB
  _Float16* qw_h     = carve((size_t)OUTD_ * OUTD_);
  _Float16* kw_h     = carve((size_t)OUTD_ * IND_);
  _Float16* vw_h     = carve((size_t)OUTD_ * IND_);
  _Float16* pw_h     = carve((size_t)OUTD_ * OUTD_);

  float* probs = (float*)d_out;                                  // [B*H][N2][N1]
  float* outp  = probs + (size_t)B_ * H_ * N2_ * N1_;            // [B,N2,OUT]

  // 1) weights -> f16
  f32_to_f16_kernel<<<(OUTD_ * OUTD_) / 256, 256, 0, stream>>>(q_w, qw_h, OUTD_ * OUTD_);
  f32_to_f16_kernel<<<(OUTD_ * IND_) / 256, 256, 0, stream>>>(k_w, kw_h, OUTD_ * IND_);
  f32_to_f16_kernel<<<(OUTD_ * IND_) / 256, 256, 0, stream>>>(v_w, vw_h, OUTD_ * IND_);
  f32_to_f16_kernel<<<(OUTD_ * OUTD_) / 256, 256, 0, stream>>>(proj_w, pw_h, OUTD_ * OUTD_);

  // 2) layernorms
  layernorm_kernel<<<B_ * N1_, 256, 0, stream>>>(in_x, ln_in_g, ln_in_b, ln_in_h, IND_);
  layernorm_kernel<<<B_ * N2_, 256, 0, stream>>>(out_x, ln_out_g, ln_out_b, ln_out_h, OUTD_);

  // 3) Q/K/V projections (WMMA + async LDS staging)
  gemm_bias_kernel<0><<<dim3(B_ * N2_ / BM, OUTD_ / BN), 256, 0, stream>>>(
      ln_out_h, qw_h, q_b, nullptr, q_h, nullptr, B_ * N2_, OUTD_, OUTD_, N2_);
  gemm_bias_kernel<0><<<dim3(B_ * N1_ / BM, OUTD_ / BN), 256, 0, stream>>>(
      ln_in_h, kw_h, k_b, nullptr, k_h, nullptr, B_ * N1_, IND_, OUTD_, N1_);
  gemm_bias_kernel<1><<<dim3(B_ * N1_ / BM, OUTD_ / BN), 256, 0, stream>>>(
      ln_in_h, vw_h, v_b, nullptr, vt_h, nullptr, B_ * N1_, IND_, OUTD_, N1_);

  // 4) scores = QK^T / sqrt(DH)  (WMMA, direct global fragments)
  attn_scores_kernel<<<dim3(N2_ / 16, N1_ / 512, B_ * H_), 256, 0, stream>>>(q_h, k_h, probs);

  // 5) softmax in place in d_out
  softmax_kernel<<<B_ * H_ * N2_, 256, 0, stream>>>(probs);

  // 6) ctrl = P V (WMMA; probs read f32 and converted in-register)
  attn_av_kernel<<<dim3(N2_ / 128, 1, B_ * H_), 256, 0, stream>>>(probs, vt_h, ctrl_h);

  // 7) out = output + ctrl @ proj_w^T + proj_b  (WMMA, f32 out + residual)
  gemm_bias_kernel<2><<<dim3(B_ * N2_ / BM, OUTD_ / BN), 256, 0, stream>>>(
      ctrl_h, pw_h, proj_b, out_x, nullptr, outp, B_ * N2_, OUTD_, OUTD_, N2_);

  (void)in_sizes; (void)n_in; (void)out_size; (void)ws_size;
}
